// CyclicShuffle_46608985096495
// MI455X (gfx1250) — 8-bit matrix operand order repaired and checked by reading, compiled, not run on hardware
//
#include <hip/hip_runtime.h>
#include <hip/hip_bf16.h>


typedef __attribute__((ext_vector_type(8))) int v8i;

#define GROUPS 4
#define CG     256
#define CTOT   1024
#define HW     784
#define NPIX   50176
#define BM     128
#define BN     64
#define BK     64
#define BN_EPS 1e-5f

__global__ void wprep_kernel(const float* __restrict__ weight,
                             const float* __restrict__ lWp,
                             const float* __restrict__ uWp,
                             unsigned* __restrict__ wsw) {
    const float lW  = lWp[0];
    const float inv = 15.0f / (uWp[0] - lW);

    unsigned widx = blockIdx.x * 256 + threadIdx.x;
    unsigned j  = widx & 7;
    unsigned l  = (widx >> 3) & 31;
    unsigned mt = (widx >> 8) & 15;
    unsigned ks = (widx >> 12) & 3;
    unsigned g  = (widx >> 14) & 3;

    unsigned m  = mt * 16 + (l & 15);
    unsigned kb = (j >> 2) * 32 + (l >> 4) * 16 + (j & 3) * 4;

    unsigned word = 0;
#pragma unroll
    for (int bb = 0; bb < 4; ++bb) {
        unsigned kg = ks * BK + kb + (unsigned)bb;
        float w = weight[(size_t)(g * CG + m) * CG + kg];
        float t = (w - lW) * inv;
        t = fminf(fmaxf(t, 0.0f), 15.0f);
        int q = (int)__builtin_rintf(t);
        int s = 2 * q - 15;
        word |= ((unsigned)(s & 0xFF)) << (8 * bb);
    }
    wsw[widx] = word;
}

__device__ __forceinline__ void quant_store(unsigned* __restrict__ dst,
                                            const float* __restrict__ r,
                                            float lA, float invA,
                                            unsigned n_loc, unsigned kq) {
#pragma unroll
    for (int c4 = 0; c4 < 4; ++c4) {
        unsigned word = 0;
#pragma unroll
        for (int bb = 0; bb < 4; ++bb) {
            float t = (r[c4 * 4 + bb] - lA) * invA;
            t = fminf(fmaxf(t, 0.0f), 15.0f);
            unsigned q = (unsigned)__builtin_rintf(t);
            word |= q << (8 * bb);
        }
        dst[n_loc * 16 + kq * 4 + c4] = word;
    }
}

__global__ __launch_bounds__(256)
void qconv_kernel(const float* __restrict__ x,
                  const unsigned* __restrict__ wsw,
                  const float* __restrict__ lAp, const float* __restrict__ uAp,
                  const float* __restrict__ gamma, const float* __restrict__ beta,
                  const float* __restrict__ rmean, const float* __restrict__ rvar,
                  float* __restrict__ out) {
    __shared__ __align__(16) unsigned lds_b[2][BN * BK / 4];
    __shared__ float lds_inv[BM];
    __shared__ float lds_bias[BM];

    const unsigned tid    = threadIdx.x;
    const unsigned lane   = tid & 31;
    const unsigned half   = lane >> 4;
    const unsigned wv     = tid >> 5;
    const unsigned wave_m = wv & 3;
    const unsigned wave_n = wv >> 2;

    const unsigned g      = blockIdx.z;
    const unsigned gs     = (g + 1) & 3;
    const unsigned base_m = blockIdx.x * BM;
    const unsigned base_n = blockIdx.y * BN;

    const float lA   = lAp[0];
    const float invA = 15.0f / (uAp[0] - lA);

    if (tid < BM) {
        unsigned c = g * CG + base_m + tid;
        float iv = gamma[c] * rsqrtf(rvar[c] + BN_EPS);
        lds_inv[tid]  = iv;
        lds_bias[tid] = beta[c] - rmean[c] * iv;
    }

    const unsigned n_loc = tid & 63;
    const unsigned kq    = tid >> 6;
    const unsigned n     = base_n + n_loc;
    const unsigned bidx  = n / HW;
    const unsigned hw    = n - bidx * HW;
    const float* src = x + (size_t)(bidx * CTOT + gs * CG) * HW + hw
                         + (size_t)(kq * 16) * HW;

    v8i acc[2][2] = {};

    {
        float r[16];
#pragma unroll
        for (int i = 0; i < 16; ++i) r[i] = src[(size_t)i * HW];
        quant_store(lds_b[0], r, lA, invA, n_loc, kq);
    }

#pragma unroll
    for (unsigned ks = 0; ks < 4; ++ks) {
        float nxt[16];
        if (ks < 3) {
            const float* s1 = src + (size_t)((ks + 1) * BK) * HW;
#pragma unroll
            for (int i = 0; i < 16; ++i) nxt[i] = s1[(size_t)i * HW];
        }

        __syncthreads();

        const unsigned* buf = lds_b[ks & 1];

        v8i bfrag[2];
#pragma unroll
        for (int tn = 0; tn < 2; ++tn) {
            unsigned nl = wave_n * 32 + (unsigned)tn * 16 + (lane & 15);
            const uint4* p = (const uint4*)&buf[nl * 16 + half * 4];
            union { uint4 q[2]; v8i v; } u;
            u.q[0] = p[0];
            u.q[1] = p[2];
            bfrag[tn] = u.v;
        }

#pragma unroll
        for (int tm = 0; tm < 2; ++tm) {
            unsigned mt = blockIdx.x * 8 + wave_m * 2 + (unsigned)tm;
            const v8i* ap =
                (const v8i*)(wsw + ((((g * 4 + ks) * 16 + mt) * 32) + lane) * 8);
            v8i afrag = *ap;
#pragma unroll
            for (int tn = 0; tn < 2; ++tn) {
                acc[tm][tn] = __builtin_amdgcn_wmma_i32_16x16x64_iu8(
                    true, afrag, false, bfrag[tn], acc[tm][tn], false, false);
            }
        }

        if (ks < 3)
            quant_store(lds_b[(ks + 1) & 1], nxt, lA, invA, n_loc, kq);
    }

#pragma unroll
    for (int tn = 0; tn < 2; ++tn) {
        unsigned nn  = base_n + wave_n * 32 + (unsigned)tn * 16 + (lane & 15);
        unsigned bo  = nn / HW;
        unsigned hwo = nn - bo * HW;
#pragma unroll
        for (int tm = 0; tm < 2; ++tm) {
#pragma unroll
            for (int r = 0; r < 8; ++r) {
                unsigned rm = wave_m * 32 + (unsigned)tm * 16 + (unsigned)r + 8 * half;
                unsigned c  = g * CG + base_m + rm;
                size_t addr = (size_t)(bo * CTOT + c) * HW + hwo;
                float f = (float)acc[tm][tn][r] * (1.0f / 225.0f);
                f = f * lds_inv[rm] + lds_bias[rm];
                f = fmaxf(f, 0.0f);
                __builtin_nontemporal_store(f + x[addr], &out[addr]);
            }
        }
    }
}

extern "C" void kernel_launch(void* const* d_in, const int* in_sizes, int n_in,
                              void* d_out, int out_size, void* d_ws, size_t ws_size,
                              hipStream_t stream) {
    const float* x      = (const float*)d_in[0];
    const float* weight = (const float*)d_in[1];
    const float* lW     = (const float*)d_in[2];
    const float* uW     = (const float*)d_in[3];
    const float* lA     = (const float*)d_in[4];
    const float* uA     = (const float*)d_in[5];
    const float* gamma  = (const float*)d_in[6];
    const float* beta   = (const float*)d_in[7];
    const float* rmean  = (const float*)d_in[8];
    const float* rvar   = (const float*)d_in[9];
    float* out          = (float*)d_out;
    unsigned* wsw       = (unsigned*)d_ws;

    wprep_kernel<<<256, 256, 0, stream>>>(weight, lW, uW, wsw);

    dim3 grid(CG / BM, NPIX / BN, GROUPS);
    qconv_kernel<<<grid, 256, 0, stream>>>(x, wsw, lA, uA,
                                           gamma, beta, rmean, rvar, out);
}
